// LSTM_37374805410197
// MI455X (gfx1250) — matrix operand order repaired and checked by reading, not compiled, not run on hardware
//
#include <hip/hip_runtime.h>
#include <hip/hip_bf16.h>
#include <math.h>

typedef __attribute__((ext_vector_type(16))) __bf16 v16bf;
typedef __attribute__((ext_vector_type(8)))  __bf16 v8bf;
typedef __attribute__((ext_vector_type(4)))  __bf16 v4bf;
typedef __attribute__((ext_vector_type(8)))  float  v8f;
typedef __attribute__((ext_vector_type(4)))  float  v4f;

// Problem sizes
constexpr int BB = 512;      // batch (M)
constexpr int HH = 2048;     // hidden
constexpr int ED = 1024;     // embed
constexpr int VV = 50257;    // vocab

// GEMM tiling: block 64x256, k-step 32, 8 wave32 each computing 32x64
constexpr int BM = 64;
constexpr int BN = 256;
constexpr int BK = 32;
constexpr int SA = 40;                 // padded LDS row stride (bf16) = 80B
constexpr int SB = 40;
constexpr int LDSA = BM * SA;          // 2560 bf16 = 5120B per buffer
constexpr int LDSB = BN * SB;          // 10240 bf16 = 20480B per buffer

union AFrag { v16bf v; struct { v8bf lo, hi; } p; };

// --- CDNA5 async global->LDS copy (16B per lane) + wait ---------------------
__device__ __forceinline__ void async_copy_b128(const void* gsrc, unsigned lds_off) {
  asm volatile("global_load_async_to_lds_b128 %0, %1, off"
               :: "v"(lds_off), "v"(gsrc) : "memory");
}
__device__ __forceinline__ void wait_async0() {
  asm volatile("s_wait_asynccnt 0" ::: "memory");
}

// ---------------------------------------------------------------------------
// One K-phase of C += A[M,K] @ W[N,K]^T with software pipelining.
// A bf16 (async-copied to LDS), W fp32 (reg-staged + converted to bf16).
// ---------------------------------------------------------------------------
__device__ __forceinline__
void gemm_phase(const __bf16* __restrict__ Ap, const float* __restrict__ Wp, int K,
                __bf16* Atile, __bf16* Btile, v8f acc[2][4],
                int tid, int bm, int bn, int N) {
  const int lane = tid & 31;
  const int wave = tid >> 5;
  const int wm   = wave & 1;        // 2 m-subtiles of 32 rows
  const int wn   = wave >> 1;       // 4 n-slabs of 64 cols
  const int hl   = lane >> 4;
  const int lm   = lane & 15;

  // ---- A staging: 64x32 bf16 = 4KB -> one async b128 per thread
  const int arow = tid >> 2;            // 0..63
  const int akk  = (tid & 3) * 8;       // 0,8,16,24
  const __bf16* agp = Ap + (size_t)(bm + arow) * K + akk;
  const unsigned alds0 = (unsigned)(uintptr_t)(Atile + arow * SA + akk);

  // ---- B staging: 256x32 fp32 -> 8 float4 per thread (rows clamped to N-1)
  const float* bgp[8];
  __bf16*      bst[8];
#pragma unroll
  for (int i = 0; i < 8; ++i) {
    const int idx = tid + i * 256;
    const int row = idx >> 3;           // 0..255
    const int kk  = (idx & 7) * 4;      // 0..28
    const int n   = bn + row < N ? bn + row : N - 1;
    bgp[i] = Wp + (size_t)n * K + kk;
    bst[i] = Btile + row * SB + kk;
  }

  const int nst = K / BK;

  // ---- prologue: issue tile 0
  async_copy_b128(agp, alds0);
  agp += BK;
  v4f wreg[8];
#pragma unroll
  for (int i = 0; i < 8; ++i) { wreg[i] = *(const v4f*)bgp[i]; bgp[i] += BK; }

  for (int s = 0; s < nst; ++s) {
    const int cur = s & 1;
    const int nxt = cur ^ 1;

    // store staged B into buf[cur] (fp32 -> bf16)
#pragma unroll
    for (int i = 0; i < 8; ++i) {
      v4bf wb;
      wb[0] = (__bf16)wreg[i][0]; wb[1] = (__bf16)wreg[i][1];
      wb[2] = (__bf16)wreg[i][2]; wb[3] = (__bf16)wreg[i][3];
      *(v4bf*)(bst[i] + cur * LDSB) = wb;
    }
    wait_async0();                 // A tile for step s resident in buf[cur]
    __syncthreads();

    // issue next tile's loads; they overlap the WMMA compute below
    if (s + 1 < nst) {
      async_copy_b128(agp, alds0 + nxt * (unsigned)(LDSA * sizeof(__bf16)));
      agp += BK;
#pragma unroll
      for (int i = 0; i < 8; ++i) { wreg[i] = *(const v4f*)bgp[i]; bgp[i] += BK; }
    }

    // ---- compute from buf[cur]
    const __bf16* Ab = Atile + cur * LDSA;
    const __bf16* Bb = Btile + cur * LDSB;

    AFrag af[2];
#pragma unroll
    for (int u = 0; u < 2; ++u) {
      const __bf16* ar = Ab + (wm * 32 + u * 16 + lm) * SA;
      af[u].p.lo = *(const v8bf*)(ar + hl * 8);
      af[u].p.hi = *(const v8bf*)(ar + 16 + hl * 8);
    }
#pragma unroll
    for (int t = 0; t < 4; ++t) {
      AFrag bf;
      const __bf16* br = Bb + (wn * 64 + t * 16 + lm) * SB;
      bf.p.lo = *(const v8bf*)(br + hl * 8);
      bf.p.hi = *(const v8bf*)(br + 16 + hl * 8);
#pragma unroll
      for (int u = 0; u < 2; ++u)
        acc[u][t] = __builtin_amdgcn_wmma_f32_16x16x32_bf16(
            false, af[u].v, false, bf.v, (short)0, acc[u][t], false, false);
    }
  }
}

// ---------------------------------------------------------------------------
// C[M,N] = A1@W1^T (+ A2@W2^T) + bias.  grid=(M/BM, ceil(N/BN)), block=256.
// grid.x (m) varies fastest so blocks sharing a W column-tile run adjacently
// and hit the 192MB L2; W streams from HBM ~once.
// ---------------------------------------------------------------------------
__global__ __launch_bounds__(256)
void gemm_bf16_wmma_kernel(const __bf16* __restrict__ A1,
                           const float*  __restrict__ W1, int K1,
                           const __bf16* __restrict__ A2,
                           const float*  __restrict__ W2, int K2,
                           const float*  __restrict__ bias,
                           float* __restrict__ C, int N) {
  __shared__ __bf16 Atile[2 * LDSA];
  __shared__ __bf16 Btile[2 * LDSB];

  const int tid = threadIdx.x;
  const int bm  = blockIdx.x * BM;
  const int bn  = blockIdx.y * BN;

  v8f acc[2][4] = {};

  gemm_phase(A1, W1, K1, Atile, Btile, acc, tid, bm, bn, N);
  if (K2 > 0)
    gemm_phase(A2, W2, K2, Atile, Btile, acc, tid, bm, bn, N);

  // ---- epilogue: lane holds col n=lm, rows hl*8+j of each 16x16 tile
  const int lane = tid & 31;
  const int wave = tid >> 5;
  const int wm = wave & 1, wn = wave >> 1;
  const int hl = lane >> 4, lm = lane & 15;
#pragma unroll
  for (int t = 0; t < 4; ++t) {
    const int n = bn + wn * 64 + t * 16 + lm;
    if (n < N) {
      const float bv = bias[n];
#pragma unroll
      for (int u = 0; u < 2; ++u) {
#pragma unroll
        for (int j = 0; j < 8; ++j) {
          const int m = bm + wm * 32 + u * 16 + hl * 8 + j;
          C[(size_t)m * N + n] = acc[u][t][j] + bv;
        }
      }
    }
  }
}

// ---------------------------------------------------------------------------
__global__ __launch_bounds__(256)
void embed_gather_kernel(const int* __restrict__ inp,
                         const float* __restrict__ E,
                         __bf16* __restrict__ xb) {
  const int b = blockIdx.x;
  const int e = threadIdx.x * 4;
  v4f v = *(const v4f*)(E + (size_t)inp[b] * ED + e);
  v4bf o; o[0]=(__bf16)v[0]; o[1]=(__bf16)v[1]; o[2]=(__bf16)v[2]; o[3]=(__bf16)v[3];
  *(v4bf*)(xb + (size_t)b * ED + e) = o;
}

__global__ __launch_bounds__(256)
void cvt_bf16_kernel(const float* __restrict__ in, __bf16* __restrict__ out) {
  const size_t i = ((size_t)blockIdx.x * 256 + threadIdx.x) * 4;
  v4f v = *(const v4f*)(in + i);
  v4bf o; o[0]=(__bf16)v[0]; o[1]=(__bf16)v[1]; o[2]=(__bf16)v[2]; o[3]=(__bf16)v[3];
  *(v4bf*)(out + i) = o;
}

__global__ __launch_bounds__(256)
void bias_comb_kernel(const float* __restrict__ bx, const float* __restrict__ bh,
                      float* __restrict__ bc) {
  const int i = blockIdx.x * 256 + threadIdx.x;
  bc[i] = bx[i] + bh[i];
}

__device__ __forceinline__ float sigm(float x) { return 1.f / (1.f + __expf(-x)); }

__global__ __launch_bounds__(256)
void lstm_pointwise_kernel(const float* __restrict__ gates,
                           const float* __restrict__ c0,
                           float* __restrict__ hx, float* __restrict__ cx,
                           __bf16* __restrict__ hxb) {
  const size_t i = ((size_t)blockIdx.x * 256 + threadIdx.x) * 4;
  const int b = (int)(i >> 11);
  const int h = (int)(i & (HH - 1));
  const float* gb = gates + (size_t)b * 4 * HH;
  v4f gf = *(const v4f*)(gb + h);
  v4f gi = *(const v4f*)(gb + HH + h);
  v4f go = *(const v4f*)(gb + 2 * HH + h);
  v4f gc = *(const v4f*)(gb + 3 * HH + h);
  v4f c  = *(const v4f*)(c0 + i);
  v4f co, ho; v4bf hb;
#pragma unroll
  for (int j = 0; j < 4; ++j) {
    const float f  = sigm(gf[j]);
    const float ig = sigm(gi[j]);
    const float o  = sigm(go[j]);
    const float ct = tanhf(gc[j]);
    const float cn = f * c[j] + ig * ct;
    const float hn = o * tanhf(cn);
    co[j] = cn; ho[j] = hn; hb[j] = (__bf16)hn;
  }
  *(v4f*)(cx + i) = co;
  *(v4f*)(hx + i) = ho;
  *(v4bf*)(hxb + i) = hb;
}

// ---------------------------------------------------------------------------
extern "C" void kernel_launch(void* const* d_in, const int* in_sizes, int n_in,
                              void* d_out, int out_size, void* d_ws, size_t ws_size,
                              hipStream_t stream) {
  const int*   inp = (const int*)  d_in[0];
  const float* h0  = (const float*)d_in[1];
  const float* c0  = (const float*)d_in[2];
  const float* E   = (const float*)d_in[3];
  const float* Wx  = (const float*)d_in[4];   // [4,H,E] -> [8192,1024]
  const float* bx  = (const float*)d_in[5];
  const float* Wh  = (const float*)d_in[6];   // [4,H,H] -> [8192,2048]
  const float* bh  = (const float*)d_in[7];
  const float* Wd  = (const float*)d_in[8];   // [V,H]
  const float* bd  = (const float*)d_in[9];

  float* out = (float*)d_out;                      // [B,V]
  float* hx  = out + (size_t)BB * VV;              // [B,H]
  float* cx  = hx + (size_t)BB * HH;               // [B,H]

  char* ws = (char*)d_ws;
  __bf16* x_bf  = (__bf16*)(ws);                              // 1MB
  __bf16* h0_bf = (__bf16*)(ws + (1u << 20));                 // 2MB
  __bf16* hx_bf = (__bf16*)(ws + (3u << 20));                 // 2MB
  float*  biasc = (float*) (ws + (5u << 20));                 // 32KB
  float*  gates = (float*) (ws + (5u << 20) + (1u << 15));    // 16MB

  embed_gather_kernel<<<dim3(BB), dim3(256), 0, stream>>>(inp, E, x_bf);
  cvt_bf16_kernel<<<dim3((BB * HH) / 1024), dim3(256), 0, stream>>>(h0, h0_bf);
  bias_comb_kernel<<<dim3((4 * HH) / 256), dim3(256), 0, stream>>>(bx, bh, biasc);

  // gates = x@Wx^T + h0@Wh^T + (bx+bh)   [M=512, N=8192]
  gemm_bf16_wmma_kernel<<<dim3(BB / BM, (4 * HH) / BN), dim3(256), 0, stream>>>(
      x_bf, Wx, ED, h0_bf, Wh, HH, biasc, gates, 4 * HH);

  lstm_pointwise_kernel<<<dim3((BB * HH) / 1024), dim3(256), 0, stream>>>(
      gates, c0, hx, cx, hx_bf);

  // out = hx@Wd^T + bd   [M=512, N=50257]
  gemm_bf16_wmma_kernel<<<dim3(BB / BM, (VV + BN - 1) / BN), dim3(256), 0, stream>>>(
      hx_bf, Wd, HH, nullptr, nullptr, 0, bd, out, VV);
}
